// AttentionMs_38869454029169
// MI455X (gfx1250) — matrix operand order repaired and checked by reading, not compiled, not run on hardware
//
#include <hip/hip_runtime.h>
#include <hip/hip_bf16.h>
#include <math.h>

// ---------------------------------------------------------------------------
// MI455X (gfx1250) implementation.
// All GEMM-shaped work (1x1 convs, 5x5 conv as 25 shifted 1x1 GEMMs) runs on
// v_wmma_f32_16x16x32_bf16. Weights are pre-packed into a contiguous-k A-fragment
// layout (k = 16*(lane/16) + s, same order as B); activations are gathered as B
// fragments straight from NCHW f32 with inline bf16 conversion.
// Waves are register-blocked over M: each B fragment is loaded once and fed
// to one WMMA per M-tile (12 for qkv, 4 for the 64-out GEMMs), minimizing
// L2 activation traffic.
// ---------------------------------------------------------------------------

typedef __attribute__((ext_vector_type(16))) __bf16 v16bf;
typedef __attribute__((ext_vector_type(8)))  float  v8f;

#define HW   65536   // 256*256
#define WDIM 256

// --- B fragment: B is K x N. Lane l: column n = l&15, K-halfs are
//     16 consecutive channels starting at kt*32 + ((l>=16)?16:0).
__device__ __forceinline__ v16bf load_b_frag(const float* base, bool ok) {
  v16bf f;
#pragma unroll
  for (int h = 0; h < 16; ++h)
    f[h] = ok ? (__bf16)base[(size_t)h * HW] : (__bf16)0.0f;
  return f;
}

// --- A fragment: pre-packed, 16 bf16 per lane, 32B-aligned vector load.
__device__ __forceinline__ v16bf load_a_frag(const __bf16* pa, int tile, int lane) {
  return *(const v16bf*)(pa + ((size_t)tile * 32 + lane) * 16);
}

__device__ __forceinline__ v8f wmma_bf16(v16bf a, v16bf b, v8f c) {
  return __builtin_amdgcn_wmma_f32_16x16x32_bf16(false, a, false, b,
                                                 (short)0, c, false, false);
}

// ---------------------------------------------------------------------------
// Weight packing into A-fragment layout.
// A tile (mt,kt) is 16x32. Lane l<16: M=l, K = kt*32 + {0..15};
// lane l>=16: M=l-16, K = kt*32 + {16..31}. Half h -> K=h (+base).
// Tiles: qkv 12mt x 2kt = 24; m1 (BN-folded) 8; m2 25 taps x 8 = 200; proj 8.
// ---------------------------------------------------------------------------
__global__ void prep_kernel(const float* __restrict__ w_qkv,
                            const float* __restrict__ w_m1,
                            const float* __restrict__ w_m2,
                            const float* __restrict__ w_proj,
                            const float* __restrict__ gamma,
                            const float* __restrict__ beta,
                            const float* __restrict__ mean,
                            const float* __restrict__ var,
                            __bf16* __restrict__ Aqkv, __bf16* __restrict__ Am1,
                            __bf16* __restrict__ Am2,  __bf16* __restrict__ Aproj,
                            float* __restrict__ b1eff) {
  int tid = blockIdx.x * blockDim.x + threadIdx.x;
  const int total = 240 * 32;
  if (tid < total) {
    int lane = tid & 31;
    int tile = tid >> 5;
    const float* w; __bf16* dst; int mt, kt, M, K; bool useScale = false;
    int tap = 0; bool isTap = false;
    if (tile < 24) {                                   // qkv 192x64
      w = w_qkv; dst = Aqkv + (size_t)tile * 512;
      mt = tile >> 1; kt = tile & 1; M = 192; K = 64;
    } else if (tile < 32) {                            // m1 64x64, BN folded
      int t2 = tile - 24; w = w_m1; dst = Am1 + (size_t)t2 * 512;
      mt = t2 >> 1; kt = t2 & 1; M = 64; K = 64; useScale = true;
    } else if (tile < 232) {                           // m2: 25 taps of 64x64
      int t2 = tile - 32; tap = t2 >> 3; int r = t2 & 7;
      mt = r >> 1; kt = r & 1; M = 64; K = 64;
      w = w_m2; dst = Am2 + (size_t)t2 * 512; isTap = true;
    } else {                                           // proj 64x64
      int t2 = tile - 232; w = w_proj; dst = Aproj + (size_t)t2 * 512;
      mt = t2 >> 1; kt = t2 & 1; M = 64; K = 64;
    }
    int mm    = mt * 16 + (lane & 15);
    int kbase = kt * 32 + ((lane >= 16) ? 16 : 0);
#pragma unroll
    for (int h = 0; h < 16; ++h) {
      int kk = kbase + h;
      float v = 0.0f;
      if (mm < M && kk < K) {
        v = isTap ? w[((size_t)mm * 64 + kk) * 25 + tap] : w[(size_t)mm * K + kk];
        if (useScale) v *= gamma[mm] * rsqrtf(var[mm] + 1e-5f);
      }
      dst[(size_t)lane * 16 + h] = (__bf16)v;
    }
  } else if (tid < total + 64) {
    int oc = tid - total;
    float s = gamma[oc] * rsqrtf(var[oc] + 1e-5f);
    b1eff[oc] = beta[oc] - mean[oc] * s;
  }
}

// ---------------------------------------------------------------------------
// qkv 1x1 conv: D[192 x 262144] = Wqkv[192x64] * X[64x262144] + bias.
// Block = 8 waves; each wave owns one 16-pixel column strip and ALL 12
// M-tiles (96 accumulator VGPRs): each B fragment feeds 12 WMMAs.
// ---------------------------------------------------------------------------
__global__ void gemm_qkv_kernel(const float* __restrict__ x,
                                const float* __restrict__ b_qkv,
                                const __bf16* __restrict__ Aqkv,
                                float* __restrict__ qkvbuf) {
  int lane = threadIdx.x & 31, wave = threadIdx.x >> 5;
  int bid = blockIdx.x;
  int ng = bid & 511; int b = bid >> 9;
  int nt = ng * 8 + wave;
  int p  = nt * 16 + (lane & 15);
  const float* xb = x + (size_t)b * 64 * HW;
  int cb = (lane >> 4) << 4;                  // channel base for this half-wave
  __builtin_prefetch(xb + (size_t)cb * HW + p, 0, 0);
  v8f acc[12];
#pragma unroll
  for (int mt = 0; mt < 12; ++mt) acc[mt] = (v8f){};
#pragma unroll
  for (int kt = 0; kt < 2; ++kt) {
    v16bf bb = load_b_frag(xb + (size_t)(kt * 32 + cb) * HW + p, true);
#pragma unroll
    for (int mt = 0; mt < 12; ++mt) {
      v16bf a = load_a_frag(Aqkv, mt * 2 + kt, lane);
      acc[mt] = wmma_bf16(a, bb, acc[mt]);
    }
  }
  int mhalf = (lane >> 4) << 3;
#pragma unroll
  for (int mt = 0; mt < 12; ++mt)
#pragma unroll
    for (int r = 0; r < 8; ++r) {
      int m = mt * 16 + r + mhalf;
      qkvbuf[((size_t)b * 192 + m) * HW + p] = acc[mt][r] + b_qkv[m];
    }
}

// ---------------------------------------------------------------------------
// Generic 64x64 1x1-conv GEMM (mask path conv1, project_out).
// One wave = 16 pixels x all 4 M-tiles; B fragments loaded once.
// ---------------------------------------------------------------------------
__global__ void gemm64_kernel(const float* __restrict__ src,
                              const float* __restrict__ bias,
                              const __bf16* __restrict__ A,
                              float* __restrict__ dst) {
  int lane = threadIdx.x & 31, wave = threadIdx.x >> 5;
  int bid = blockIdx.x;
  int ng = bid & 511; int b = bid >> 9;
  int nt = ng * 8 + wave;
  int p  = nt * 16 + (lane & 15);
  const float* sb = src + (size_t)b * 64 * HW;
  int cb = (lane >> 4) << 4;
  __builtin_prefetch(sb + (size_t)cb * HW + p, 0, 0);
  v8f acc[4];
#pragma unroll
  for (int mt = 0; mt < 4; ++mt) acc[mt] = (v8f){};
#pragma unroll
  for (int kt = 0; kt < 2; ++kt) {
    v16bf bb = load_b_frag(sb + (size_t)(kt * 32 + cb) * HW + p, true);
#pragma unroll
    for (int mt = 0; mt < 4; ++mt) {
      v16bf a = load_a_frag(A, mt * 2 + kt, lane);
      acc[mt] = wmma_bf16(a, bb, acc[mt]);
    }
  }
  int mhalf = (lane >> 4) << 3;
#pragma unroll
  for (int mt = 0; mt < 4; ++mt)
#pragma unroll
    for (int r = 0; r < 8; ++r) {
      int m = mt * 16 + r + mhalf;
      dst[((size_t)b * 64 + m) * HW + p] = acc[mt][r] + bias[m];
    }
}

// ---------------------------------------------------------------------------
// Depthwise 3x3 (+bias) on the first 128 qkv channels (q,k). Bandwidth-bound.
// ---------------------------------------------------------------------------
__global__ void dwconv3_kernel(const float* __restrict__ qkvbuf,
                               const float* __restrict__ w_dw,
                               const float* __restrict__ b_dw,
                               float* __restrict__ qkbuf) {
  size_t idx = (size_t)blockIdx.x * blockDim.x + threadIdx.x;   // 4*128*65536
  int p  = (int)(idx & (HW - 1));
  int ch = (int)((idx >> 16) & 127);
  int b  = (int)(idx >> 23);
  int r = p >> 8, cc = p & 255;
  const float* src = qkvbuf + ((size_t)b * 192 + ch) * HW;
  float acc = b_dw[ch];
#pragma unroll
  for (int dy = -1; dy <= 1; ++dy)
#pragma unroll
    for (int dx = -1; dx <= 1; ++dx) {
      int rr = r + dy, c2 = cc + dx;
      if ((unsigned)rr < 256u && (unsigned)c2 < 256u)
        acc += w_dw[ch * 9 + (dy + 1) * 3 + (dx + 1)] * src[rr * WDIM + c2];
    }
  qkbuf[((size_t)b * 128 + ch) * HW + p] = acc;
}

// ---------------------------------------------------------------------------
// 5x5 conv (pad 2, no bias) as 25 shifted 1x1 GEMMs + fused sigmoid gate:
// v = x * (1 + sigmoid(conv)). One wave = 16 pixels x all 4 M-tiles;
// each B fragment (25 taps x 2 k-steps) feeds 4 WMMAs -> 200 WMMAs/wave.
// ---------------------------------------------------------------------------
__global__ void conv5x5_gate_kernel(const float* __restrict__ ybuf,
                                    const float* __restrict__ x,
                                    const __bf16* __restrict__ Am2,
                                    float* __restrict__ vbuf) {
  int lane = threadIdx.x & 31, wave = threadIdx.x >> 5;
  int bid = blockIdx.x;
  int ng = bid & 511; int b = bid >> 9;
  int nt = ng * 8 + wave;
  int p  = nt * 16 + (lane & 15);
  int r = p >> 8, cc = p & 255;
  int cb = (lane >> 4) << 4;
  const float* yb = ybuf + (size_t)b * 64 * HW;
  v8f acc[4];
#pragma unroll
  for (int mt = 0; mt < 4; ++mt) acc[mt] = (v8f){};
#pragma unroll 1
  for (int t = 0; t < 25; ++t) {
    int dy = t / 5 - 2, dx = t % 5 - 2;
    int rr = r + dy, c2 = cc + dx;
    bool ok = ((unsigned)rr < 256u) && ((unsigned)c2 < 256u);
    const float* base0 = yb + (size_t)cb * HW + rr * WDIM + c2;
#pragma unroll
    for (int kt = 0; kt < 2; ++kt) {
      v16bf bb = load_b_frag(base0 + (size_t)(kt * 32) * HW, ok);
#pragma unroll
      for (int mt = 0; mt < 4; ++mt) {
        v16bf a = load_a_frag(Am2, t * 8 + mt * 2 + kt, lane);
        acc[mt] = wmma_bf16(a, bb, acc[mt]);
      }
    }
  }
  int mhalf = (lane >> 4) << 3;
#pragma unroll
  for (int mt = 0; mt < 4; ++mt)
#pragma unroll
    for (int rI = 0; rI < 8; ++rI) {
      int m = mt * 16 + rI + mhalf;
      size_t o = ((size_t)b * 64 + m) * HW + p;
      float xv  = x[o];
      float sig = 1.0f / (1.0f + __expf(-acc[mt][rI]));
      vbuf[o] = xv * (1.0f + sig);
    }
}

// ---------------------------------------------------------------------------
// Per-(b,channel) L2 norms over n=65536 for q (rows 0..63) and k (64..127).
// ---------------------------------------------------------------------------
__global__ void rownorm_kernel(const float* __restrict__ qkbuf,
                               float* __restrict__ norms) {
  __shared__ float sdata[256];
  int row = blockIdx.x;                 // 512 rows
  const float* src = qkbuf + (size_t)row * HW;
  float s = 0.0f;
  for (int n = threadIdx.x; n < HW; n += 256) { float v = src[n]; s += v * v; }
  sdata[threadIdx.x] = s; __syncthreads();
  for (int st = 128; st > 0; st >>= 1) {
    if ((int)threadIdx.x < st) sdata[threadIdx.x] += sdata[threadIdx.x + st];
    __syncthreads();
  }
  if (threadIdx.x == 0) norms[row] = fmaxf(sqrtf(sdata[0]), 1e-12f);
}

// ---------------------------------------------------------------------------
// attn[b,h] = softmax( (q @ k^T) / (|q||k|) * temperature )   (8x8 per head)
// Norm scaling folded here so q/k never need a normalize pass.
// ---------------------------------------------------------------------------
__global__ void attn_kernel(const float* __restrict__ qkbuf,
                            const float* __restrict__ norms,
                            const float* __restrict__ temperature,
                            float* __restrict__ attnw) {
  __shared__ float red[256];
  __shared__ float Sm[64];
  int bh = blockIdx.x; int h = bh & 7; int b = bh >> 3;
  const float* qb = qkbuf + ((size_t)b * 128 + h * 8) * HW;
  const float* kb = qkbuf + ((size_t)b * 128 + 64 + h * 8) * HW;
  float S[64];
#pragma unroll
  for (int i = 0; i < 64; ++i) S[i] = 0.0f;
  for (int n = threadIdx.x; n < HW; n += 256) {
    float qv[8], kv[8];
#pragma unroll
    for (int i = 0; i < 8; ++i) qv[i] = qb[(size_t)i * HW + n];
#pragma unroll
    for (int j = 0; j < 8; ++j) kv[j] = kb[(size_t)j * HW + n];
#pragma unroll
    for (int i = 0; i < 8; ++i)
#pragma unroll
      for (int j = 0; j < 8; ++j) S[i * 8 + j] += qv[i] * kv[j];
  }
#pragma unroll 1
  for (int e = 0; e < 64; ++e) {
    red[threadIdx.x] = S[e]; __syncthreads();
    for (int st = 128; st > 0; st >>= 1) {
      if ((int)threadIdx.x < st) red[threadIdx.x] += red[threadIdx.x + st];
      __syncthreads();
    }
    if (threadIdx.x == 0) Sm[e] = red[0];
    __syncthreads();
  }
  if (threadIdx.x < 8) {
    int ci = threadIdx.x;
    float temp = temperature[h];
    float nq = norms[b * 128 + h * 8 + ci];
    float a[8]; float mx = -1e30f;
#pragma unroll
    for (int d = 0; d < 8; ++d) {
      float nk = norms[b * 128 + 64 + h * 8 + d];
      a[d] = Sm[ci * 8 + d] / (nq * nk) * temp;
      mx = fmaxf(mx, a[d]);
    }
    float sum = 0.0f;
#pragma unroll
    for (int d = 0; d < 8; ++d) { a[d] = __expf(a[d] - mx); sum += a[d]; }
    float inv = 1.0f / sum;
#pragma unroll
    for (int d = 0; d < 8; ++d) attnw[((size_t)bh * 8 + ci) * 8 + d] = a[d] * inv;
  }
}

// ---------------------------------------------------------------------------
// out[b, h*8+ci, p] = sum_d attn[b,h,ci,d] * v[b, h*8+d, p]. Bandwidth-bound.
// ---------------------------------------------------------------------------
__global__ void apply_attn_kernel(const float* __restrict__ attnw,
                                  const float* __restrict__ vbuf,
                                  float* __restrict__ obuf) {
  __shared__ float aw[512];
  size_t idx = (size_t)blockIdx.x * 256 + threadIdx.x;   // 4*65536
  int p = (int)(idx & (HW - 1));
  int b = (int)(idx >> 16);
  for (int i = threadIdx.x; i < 512; i += 256) aw[i] = attnw[(size_t)b * 512 + i];
  __syncthreads();
  const float* vb = vbuf + (size_t)b * 64 * HW;
  float* ob = obuf + (size_t)b * 64 * HW;
#pragma unroll 1
  for (int h = 0; h < 8; ++h) {
    float vv[8];
#pragma unroll
    for (int d = 0; d < 8; ++d) vv[d] = vb[(size_t)(h * 8 + d) * HW + p];
#pragma unroll
    for (int ci = 0; ci < 8; ++ci) {
      float o = 0.0f;
#pragma unroll
      for (int d = 0; d < 8; ++d) o += aw[(h * 8 + ci) * 8 + d] * vv[d];
      ob[(size_t)(h * 8 + ci) * HW + p] = o;
    }
  }
}

// ---------------------------------------------------------------------------
extern "C" void kernel_launch(void* const* d_in, const int* in_sizes, int n_in,
                              void* d_out, int out_size, void* d_ws, size_t ws_size,
                              hipStream_t stream) {
  const float* x      = (const float*)d_in[0];
  const float* w_qkv  = (const float*)d_in[1];
  const float* b_qkv  = (const float*)d_in[2];
  const float* w_dw   = (const float*)d_in[3];
  const float* b_dw   = (const float*)d_in[4];
  const float* w_proj = (const float*)d_in[5];
  const float* b_proj = (const float*)d_in[6];
  const float* temp   = (const float*)d_in[7];
  const float* w_m1   = (const float*)d_in[8];
  const float* gamma  = (const float*)d_in[9];
  const float* beta   = (const float*)d_in[10];
  const float* mean   = (const float*)d_in[11];
  const float* var    = (const float*)d_in[12];
  const float* w_m2   = (const float*)d_in[13];
  float* out = (float*)d_out;

  char* ws = (char*)d_ws;
  __bf16* Aqkv  = (__bf16*)(ws + 0);        //  24 tiles * 512 bf16 = 24576 B
  __bf16* Am1   = (__bf16*)(ws + 24576);    //   8 tiles           =  8192 B
  __bf16* Am2   = (__bf16*)(ws + 32768);    // 200 tiles           = 204800 B
  __bf16* Aproj = (__bf16*)(ws + 237568);   //   8 tiles           =  8192 B
  float*  b1eff = (float*)(ws + 245760);    //   64 f32
  float*  norms = (float*)(ws + 246016);    //  512 f32
  float*  attnw = (float*)(ws + 248064);    // 2048 f32
  float*  qkvbuf = (float*)(ws + 262144);                 // 4*192*65536 f32
  float*  qkbuf  = qkvbuf + (size_t)4 * 192 * HW;         // 4*128*65536 f32
  float*  ybuf   = qkbuf  + (size_t)4 * 128 * HW;         // 4*64*65536 f32
  float*  vbuf   = ybuf   + (size_t)4 * 64  * HW;         // 4*64*65536 f32
  float*  obuf   = vbuf   + (size_t)4 * 64  * HW;         // 4*64*65536 f32

  prep_kernel<<<31, 256, 0, stream>>>(w_qkv, w_m1, w_m2, w_proj,
                                      gamma, beta, mean, var,
                                      Aqkv, Am1, Am2, Aproj, b1eff);
  gemm_qkv_kernel<<<4 * 512, 256, 0, stream>>>(x, b_qkv, Aqkv, qkvbuf);
  gemm64_kernel<<<4 * 512, 256, 0, stream>>>(x, b1eff, Am1, ybuf);
  dwconv3_kernel<<<131072, 256, 0, stream>>>(qkvbuf, w_dw, b_dw, qkbuf);
  conv5x5_gate_kernel<<<4 * 512, 256, 0, stream>>>(ybuf, x, Am2, vbuf);
  rownorm_kernel<<<512, 256, 0, stream>>>(qkbuf, norms);
  attn_kernel<<<32, 256, 0, stream>>>(qkbuf, norms, temp, attnw);
  apply_attn_kernel<<<1024, 256, 0, stream>>>(attnw, vbuf, obuf);
  gemm64_kernel<<<4 * 512, 256, 0, stream>>>(obuf, b_proj, Aproj, out);
}
